// NCM_classifier_33698313404887
// MI455X (gfx1250) — matrix operand order repaired and checked by reading, not compiled, not run on hardware
//
#include <hip/hip_runtime.h>
#include <hip/hip_bf16.h>
#include <math.h>

typedef __attribute__((ext_vector_type(16))) _Float16 v16h;
typedef __attribute__((ext_vector_type(8)))  _Float16 v8h;
typedef __attribute__((ext_vector_type(8)))  float    v8f;

static constexpr int Bsz = 8192;
static constexpr int Csz = 2048;
static constexpr int Fsz = 2048;

// ---------------------------------------------------------------------------
// Kernel 1: xs = x / std  -> f16, and x2[row] = sum(xs*xs) in fp32
// One block (256 threads) per row.
// ---------------------------------------------------------------------------
__global__ __launch_bounds__(256)
void prep_x_kernel(const float* __restrict__ x,
                   const float* __restrict__ variance,
                   _Float16* __restrict__ xs_h,
                   float* __restrict__ x2)
{
    const int row = blockIdx.x;
    const float inv = 1.0f / variance[0];
    const float* xr = x + (size_t)row * Fsz;
    _Float16* xo = xs_h + (size_t)row * Fsz;

    float s = 0.0f;
    for (int i = threadIdx.x; i < Fsz; i += 256) {
        float v = xr[i] * inv;
        xo[i] = (_Float16)v;
        s += v * v;
    }
    __shared__ float red[256];
    red[threadIdx.x] = s;
    __syncthreads();
    for (int off = 128; off > 0; off >>= 1) {
        if (threadIdx.x < off) red[threadIdx.x] += red[threadIdx.x + off];
        __syncthreads();
    }
    if (threadIdx.x == 0) x2[row] = red[0];
}

// ---------------------------------------------------------------------------
// Kernel 2: means -> f16, and m2[row] = sum(means*means) in fp32
// ---------------------------------------------------------------------------
__global__ __launch_bounds__(256)
void prep_m_kernel(const float* __restrict__ m,
                   _Float16* __restrict__ m_h,
                   float* __restrict__ m2)
{
    const int row = blockIdx.x;
    const float* mr = m + (size_t)row * Fsz;
    _Float16* mo = m_h + (size_t)row * Fsz;

    float s = 0.0f;
    for (int i = threadIdx.x; i < Fsz; i += 256) {
        float v = mr[i];
        mo[i] = (_Float16)v;
        s += v * v;
    }
    __shared__ float red[256];
    red[threadIdx.x] = s;
    __syncthreads();
    for (int off = 128; off > 0; off >>= 1) {
        if (threadIdx.x < off) red[threadIdx.x] += red[threadIdx.x + off];
        __syncthreads();
    }
    if (threadIdx.x == 0) m2[row] = red[0];
}

// ---------------------------------------------------------------------------
// Kernel 3: WMMA GEMM + fused distance epilogue, double-buffered fragments.
//   cross[b,c] = sum_f xs[b,f] * means[c,f]
//   dist = sqrt(max(x2[b] + m2[c] - 2*cross, 0)); exponent = -dist
// Block: 256 threads = 8 waves, block tile 128(M) x 128(N).
// Wave tile: 64(M) x 32(N) = 4x2 tiles of 16x16; K-step = 32.
//
// Fragment layouts used here (wave32; A and B share contiguous K order 16*(L>>4)+s):
//  A (16x32 f16): lane L holds row M=L&15; VGPR0..3 = K in [16*(L>>4), +8),
//                 VGPR4..7 = K in [16*(L>>4)+8, +8)  -> two 16B loads.
//  B (32x16 f16): lane L holds col N=L&15; VGPR0..7 = K in [16*(L>>4), +16)
//                 -> since B = means^T tile, col N is a contiguous row of
//                 means: one 32B load.
//  C/D (16x16 f32): VGPR v: M = v + 8*(L>>4), N = L&15.
// ---------------------------------------------------------------------------

// Load fragment set BUF at K-displacement KOFF (halves) from current apk/bpk.
// All displacements (mt*16*Fsz, nt*16*Fsz, KOFF) are compile-time constants
// so they fold into the 24-bit instruction offset of global_load_b128.
#define LOAD_FRAGS(BUF, KOFF)                                                  \
    {                                                                          \
        _Pragma("unroll") for (int mt = 0; mt < 4; ++mt) {                     \
            const _Float16* ap = apk + (size_t)mt * 16 * Fsz + (KOFF);         \
            *(v8h*)&afrag[BUF][mt]       = *(const v8h*)ap;                    \
            *((v8h*)&afrag[BUF][mt] + 1) = *(const v8h*)(ap + 8);              \
        }                                                                      \
        _Pragma("unroll") for (int nt = 0; nt < 2; ++nt) {                     \
            bfrag[BUF][nt] =                                                   \
                *(const v16h*)(bpk + (size_t)nt * 16 * Fsz + (KOFF));          \
        }                                                                      \
    }

#define MMA_FRAGS(BUF)                                                         \
    {                                                                          \
        _Pragma("unroll") for (int mt = 0; mt < 4; ++mt)                       \
            _Pragma("unroll") for (int nt = 0; nt < 2; ++nt)                   \
                acc[mt][nt] = __builtin_amdgcn_wmma_f32_16x16x32_f16(          \
                    /*neg_a=*/false, afrag[BUF][mt],                           \
                    /*neg_b=*/false, bfrag[BUF][nt],                           \
                    /*c_mod=*/(short)0, acc[mt][nt],                           \
                    /*reuse_a=*/false, /*reuse_b=*/false);                     \
    }

__global__ __launch_bounds__(256)
void gemm_dist_kernel(const _Float16* __restrict__ xs,   // [B,F] f16
                      const _Float16* __restrict__ mh,   // [C,F] f16
                      const float* __restrict__ x2,      // [B]
                      const float* __restrict__ m2,      // [C]
                      float* __restrict__ out_exp,       // [B,C]
                      float* __restrict__ out_dist)      // [B,C]
{
    const int wave  = threadIdx.x >> 5;
    const int lane  = threadIdx.x & 31;
    const int waveM = wave & 1;    // 0..1
    const int waveN = wave >> 1;   // 0..3
    const int m0 = blockIdx.x * 128 + waveM * 64;  // wave rows (over B)
    const int n0 = blockIdx.y * 128 + waveN * 32;  // wave cols (over C)

    const int lrow = lane & 15;
    const int lhi  = lane >> 4;    // 0/1: K-half selector

    v8f acc[4][2] = {};
    v16h afrag[2][4];
    v16h bfrag[2][2];

    // Per-lane base pointers at K=0; advanced by 64 halves per 2 K-steps.
    const _Float16* apk = xs + (size_t)(m0 + lrow) * Fsz + 16 * lhi;
    const _Float16* bpk = mh + (size_t)(n0 + lrow) * Fsz + 16 * lhi;

    // Software pipeline: fragments for K-step i+1 are in flight while the
    // WMMAs for K-step i execute.
    LOAD_FRAGS(0, 0)
    for (int k = 0; k < Fsz - 64; k += 64) {
        LOAD_FRAGS(1, 32)
        MMA_FRAGS(0)
        LOAD_FRAGS(0, 64)
        MMA_FRAGS(1)
        apk += 64;
        bpk += 64;
    }
    // Tail: last two K-steps (no further prefetch).
    LOAD_FRAGS(1, 32)
    MMA_FRAGS(0)
    MMA_FRAGS(1)

    // Epilogue: d2 = x2 + m2 - 2*cross, clamp, sqrt; write dist and -dist.
#pragma unroll
    for (int nt = 0; nt < 2; ++nt) {
        const int c = n0 + nt * 16 + lrow;
        const float mm = m2[c];
#pragma unroll
        for (int mt = 0; mt < 4; ++mt) {
#pragma unroll
            for (int v = 0; v < 8; ++v) {
                const int r = m0 + mt * 16 + 8 * lhi + v;
                float d2 = x2[r] + mm - 2.0f * acc[mt][nt][v];
                d2 = fmaxf(d2, 0.0f);
                const float dist = __builtin_sqrtf(d2);
                const size_t idx = (size_t)r * Csz + c;
                out_dist[idx] = dist;
                out_exp[idx]  = -dist;
            }
        }
    }
}

// ---------------------------------------------------------------------------
// Kernel 4: row softmax over C. One block (256 threads) per row.
// ---------------------------------------------------------------------------
__global__ __launch_bounds__(256)
void softmax_kernel(const float* __restrict__ expo,
                    float* __restrict__ probs)
{
    const int row = blockIdx.x;
    const float* er = expo + (size_t)row * Csz;
    float* pr = probs + (size_t)row * Csz;

    __shared__ float red[256];

    float mx = -INFINITY;
    for (int i = threadIdx.x; i < Csz; i += 256) mx = fmaxf(mx, er[i]);
    red[threadIdx.x] = mx;
    __syncthreads();
    for (int off = 128; off > 0; off >>= 1) {
        if (threadIdx.x < off)
            red[threadIdx.x] = fmaxf(red[threadIdx.x], red[threadIdx.x + off]);
        __syncthreads();
    }
    mx = red[0];
    __syncthreads();

    float s = 0.0f;
    for (int i = threadIdx.x; i < Csz; i += 256) s += __expf(er[i] - mx);
    red[threadIdx.x] = s;
    __syncthreads();
    for (int off = 128; off > 0; off >>= 1) {
        if (threadIdx.x < off) red[threadIdx.x] += red[threadIdx.x + off];
        __syncthreads();
    }
    const float inv = 1.0f / red[0];

    for (int i = threadIdx.x; i < Csz; i += 256)
        pr[i] = __expf(er[i] - mx) * inv;
}

// ---------------------------------------------------------------------------
// Launch
// ---------------------------------------------------------------------------
extern "C" void kernel_launch(void* const* d_in, const int* in_sizes, int n_in,
                              void* d_out, int out_size, void* d_ws, size_t ws_size,
                              hipStream_t stream) {
    const float* x        = (const float*)d_in[0];  // [B,F]
    const float* means    = (const float*)d_in[1];  // [C,F]
    const float* variance = (const float*)d_in[2];  // [1]

    float* out = (float*)d_out;
    const size_t BC = (size_t)Bsz * Csz;
    float* probs = out;            // output 0
    float* expo  = out + BC;       // output 1
    float* dist  = out + 2 * BC;   // output 2

    // Workspace layout: xs_h [B*F] f16 | means_h [C*F] f16 | x2 [B] | m2 [C]
    _Float16* xs_h = (_Float16*)d_ws;
    _Float16* mh   = xs_h + (size_t)Bsz * Fsz;
    float* x2 = (float*)(mh + (size_t)Csz * Fsz);
    float* m2 = x2 + Bsz;

    prep_x_kernel<<<Bsz, 256, 0, stream>>>(x, variance, xs_h, x2);
    prep_m_kernel<<<Csz, 256, 0, stream>>>(means, mh, m2);

    dim3 grid(Bsz / 128, Csz / 128);   // 64 x 16 blocks
    gemm_dist_kernel<<<grid, 256, 0, stream>>>(xs_h, mh, x2, m2, expo, dist);

    softmax_kernel<<<Bsz, 256, 0, stream>>>(expo, probs);
}
